// SandwichTransformerBlock_2095944041029
// MI455X (gfx1250) — matrix operand order repaired and checked by reading, not compiled, not run on hardware
//
#include <hip/hip_runtime.h>

// ---------------------------------------------------------------------------
// Sandwich transformer block for MI455X (gfx1250): bf16 WMMA everywhere,
// async global->LDS staging (ASYNCcnt) with double buffering in the GEMMs.
// H=2048, NH=16, HD=128, I=8192, B=2, S=2048.
// ---------------------------------------------------------------------------

#define HDIM   2048
#define NHEADS 16
#define HEADD  128
#define IDIM   8192
#define BATCH  2
#define SEQ    2048
#define ROWS   (BATCH * SEQ)   // 4096

typedef __bf16 bf16_t;
typedef __attribute__((ext_vector_type(16))) __bf16 v16bf;
typedef __attribute__((ext_vector_type(8)))  float  v8f;

union FragB16 {
    unsigned int u[8];
    v16bf v;
};

__device__ __forceinline__ v8f v8f_zero() {
    v8f z = {0.f, 0.f, 0.f, 0.f, 0.f, 0.f, 0.f, 0.f};
    return z;
}

// --- CDNA5 async global->LDS helpers ---------------------------------------
typedef __attribute__((address_space(3))) char lds_char_t;

__device__ __forceinline__ unsigned lds_off_of(void* p) {
    // generic -> LDS addrspace -> 32-bit LDS byte offset
    return (unsigned)(unsigned long long)(lds_char_t*)p;
}

__device__ __forceinline__ void async_load_b128(unsigned lds_off, const void* gaddr) {
    // per-lane: LDS[lds_off] <= 16 bytes from gaddr; tracked by ASYNCcnt
    asm volatile("global_load_async_to_lds_b128 %0, %1, off"
                 :: "v"(lds_off), "v"(gaddr)
                 : "memory");
}

__device__ __forceinline__ void wait_async_le4() {
    asm volatile("s_wait_asynccnt 0x4" ::: "memory");
}
__device__ __forceinline__ void wait_async_0() {
    asm volatile("s_wait_asynccnt 0x0" ::: "memory");
}

// Stage a 128x32 bf16 slab (rows row0..row0+127, cols k0..k0+31 of a row-major
// [.., row_stride] matrix) into LDS at lds_base. 256 threads, 2 x 16B each.
__device__ __forceinline__ void stage_tile_async(const bf16_t* gbase, int row_stride,
                                                 int row0, int k0, unsigned lds_base,
                                                 int tid) {
#pragma unroll
    for (int jj = 0; jj < 2; ++jj) {
        int c = tid + jj * 256;      // chunk id in [0, 512): 16B chunks
        int r = c >> 2, q = c & 3;   // 4 chunks (64B) per 32-elem row
        async_load_b128(lds_base + c * 16,
                        gbase + (size_t)(row0 + r) * row_stride + k0 + q * 8);
    }
}

// ---------------------------------------------------------------------------
// fp32 -> bf16 conversion (weights / activations)
// ---------------------------------------------------------------------------
__global__ void f32_to_bf16_kernel(const float* __restrict__ src,
                                   bf16_t* __restrict__ dst, long n) {
    long i = (long)blockIdx.x * blockDim.x + threadIdx.x;
    long stride = (long)gridDim.x * blockDim.x;
    for (; i < n; i += stride) dst[i] = (bf16_t)src[i];
}

// ---------------------------------------------------------------------------
// rmsnorm chain: t = a (+ bres); y = rmsnorm(t, w1)
//   if out_f: write y (fp32)
//   if w2:    z = rmsnorm(y, w2); write z as bf16 to out_b
//   else if out_b: write y as bf16
// ---------------------------------------------------------------------------
__global__ __launch_bounds__(256) void rmsnorm_chain_kernel(
    const float* __restrict__ a, const float* __restrict__ bres,
    const float* __restrict__ w1, const float* __restrict__ w2,
    float* __restrict__ out_f, bf16_t* __restrict__ out_b) {
    __shared__ float red[256];
    const int tid = threadIdx.x;
    const size_t base = (size_t)blockIdx.x * HDIM;

    float v[8];
    float ss = 0.f;
#pragma unroll
    for (int j = 0; j < 8; ++j) {
        int c = tid + j * 256;
        float t = a[base + c];
        if (bres) t += bres[base + c];
        v[j] = t;
        ss += t * t;
    }
    red[tid] = ss;
    __syncthreads();
    for (int s = 128; s > 0; s >>= 1) {
        if (tid < s) red[tid] += red[tid + s];
        __syncthreads();
    }
    float inv = rsqrtf(red[0] * (1.0f / HDIM) + 1e-6f);

    float y[8];
    float ss2 = 0.f;
#pragma unroll
    for (int j = 0; j < 8; ++j) {
        int c = tid + j * 256;
        y[j] = v[j] * inv * w1[c];
        if (out_f) out_f[base + c] = y[j];
        ss2 += y[j] * y[j];
    }

    if (w2) {
        __syncthreads();
        red[tid] = ss2;
        __syncthreads();
        for (int s = 128; s > 0; s >>= 1) {
            if (tid < s) red[tid] += red[tid + s];
            __syncthreads();
        }
        float inv2 = rsqrtf(red[0] * (1.0f / HDIM) + 1e-6f);
#pragma unroll
        for (int j = 0; j < 8; ++j) {
            int c = tid + j * 256;
            out_b[base + c] = (bf16_t)(y[j] * inv2 * w2[c]);
        }
    } else if (out_b) {
#pragma unroll
        for (int j = 0; j < 8; ++j) {
            int c = tid + j * 256;
            out_b[base + c] = (bf16_t)y[j];
        }
    }
}

// ---------------------------------------------------------------------------
// Tiled bf16 GEMM: C[M,N] = A[M,K] * W[N,K]^T   (row-major, x @ W.T form)
// 128x128 tile per block, 8 waves (4x2), 32x64 per wave = 2x4 WMMA frags.
// K-step 32, double-buffered async global->LDS staging:
//   issue(next) -> s_wait_asynccnt 4 -> barrier -> 8 WMMA -> barrier
// ---------------------------------------------------------------------------
template <bool OUT_BF16>
__global__ __launch_bounds__(256) void gemm_bf16_kernel(
    const bf16_t* __restrict__ A, const bf16_t* __restrict__ W,
    void* __restrict__ Cout, int M, int N, int K) {
    __shared__ bf16_t As[2][128 * 32];
    __shared__ bf16_t Ws[2][128 * 32];

    const int tid  = threadIdx.x;
    const int wid  = tid >> 5;
    const int lane = tid & 31;
    const int hl   = lane >> 4;   // half-wave select
    const int l16  = lane & 15;
    const int wm   = wid >> 1;    // 0..3
    const int wn   = wid & 1;     // 0..1
    const int m_tile = blockIdx.y * 128;
    const int n_tile = blockIdx.x * 128;

    unsigned baseA[2] = { lds_off_of(&As[0][0]), lds_off_of(&As[1][0]) };
    unsigned baseW[2] = { lds_off_of(&Ws[0][0]), lds_off_of(&Ws[1][0]) };

    v8f acc[2][4];
#pragma unroll
    for (int i = 0; i < 2; ++i)
#pragma unroll
        for (int j = 0; j < 4; ++j) acc[i][j] = v8f_zero();

    // prologue: stage first K-slab into buffer 0 (4 async loads / thread)
    stage_tile_async(A, K, m_tile, 0, baseA[0], tid);
    stage_tile_async(W, K, n_tile, 0, baseW[0], tid);

    int p = 0;
    for (int k0 = 0; k0 < K; k0 += 32, p ^= 1) {
        const bool more = (k0 + 32) < K;
        if (more) {
            // overwrite p^1: safe, trailing barrier of previous iter passed
            stage_tile_async(A, K, m_tile, k0 + 32, baseA[p ^ 1], tid);
            stage_tile_async(W, K, n_tile, k0 + 32, baseW[p ^ 1], tid);
            wait_async_le4();   // oldest 4 (= current buffer) retired, in order
        } else {
            wait_async_0();
        }
        __syncthreads();

        FragB16 af[2], bf[4];
#pragma unroll
        for (int i = 0; i < 2; ++i) {
            int mrow = wm * 32 + i * 16 + l16;
#pragma unroll
            for (int v = 0; v < 8; ++v) {
                int kb = hl * 16 + 2 * v;
                af[i].u[v] = *(const unsigned int*)&As[p][mrow * 32 + kb];
            }
        }
#pragma unroll
        for (int j = 0; j < 4; ++j) {
            int nrow = wn * 64 + j * 16 + l16;
#pragma unroll
            for (int v = 0; v < 8; ++v) {
                int kb = hl * 16 + 2 * v;
                bf[j].u[v] = *(const unsigned int*)&Ws[p][nrow * 32 + kb];
            }
        }
#pragma unroll
        for (int i = 0; i < 2; ++i)
#pragma unroll
            for (int j = 0; j < 4; ++j)
                acc[i][j] = __builtin_amdgcn_wmma_f32_16x16x32_bf16(
                    false, af[i].v, false, bf[j].v, (short)0, acc[i][j], false, false);

        __syncthreads();   // all waves done reading buffer p before overwrite
    }

#pragma unroll
    for (int i = 0; i < 2; ++i)
#pragma unroll
        for (int j = 0; j < 4; ++j)
#pragma unroll
            for (int r = 0; r < 8; ++r) {
                int gm = m_tile + wm * 32 + i * 16 + r + 8 * hl;
                int gn = n_tile + wn * 64 + j * 16 + l16;
                float val = acc[i][j][r];
                if (OUT_BF16)
                    ((bf16_t*)Cout)[(size_t)gm * N + gn] = (bf16_t)val;
                else
                    ((float*)Cout)[(size_t)gm * N + gn] = val;
            }
}

// ---------------------------------------------------------------------------
// Flash attention (causal), one block per (128 q-rows, head, batch).
// 8 waves; wave w owns q-rows [16w,16w+16) and all 128 key columns of a tile.
// Q/K tiles staged with async global->LDS; V transposed element-wise.
// Dynamic LDS: Qs 32K | Ks 32K | Vt 32K | Ps 32K = 128 KB.
// ---------------------------------------------------------------------------
__global__ __launch_bounds__(256) void attn_kernel(
    const bf16_t* __restrict__ Q, const bf16_t* __restrict__ Kmat,
    const bf16_t* __restrict__ V, bf16_t* __restrict__ O) {
    extern __shared__ __align__(16) char smem[];
    bf16_t* Qs = (bf16_t*)smem;               // [128][128]  (q-row, d)
    bf16_t* Ks = (bf16_t*)(smem + 32768);     // [128][128]  (t, d)
    bf16_t* Vt = (bf16_t*)(smem + 65536);     // [128][128]  (d, t)  transposed
    bf16_t* Ps = (bf16_t*)(smem + 98304);     // [8 waves][16][128]

    const int qt = blockIdx.x;   // q tile
    const int h  = blockIdx.y;   // head
    const int b  = blockIdx.z;   // batch
    const int tid = threadIdx.x, wid = tid >> 5, lane = tid & 31;
    const int hl = lane >> 4, l16 = lane & 15;
    const size_t rowbase = (size_t)b * SEQ;
    const int col0 = h * HEADD;
    const float scale = 0.08838834764831845f;  // 1/sqrt(128)
    const float NEG_INF = -__builtin_inff();

    const unsigned qbase = lds_off_of(Qs);
    const unsigned kbase = lds_off_of(Ks);

    // stage Q tile: 2048 x 16B chunks, 8 per thread (async)
#pragma unroll
    for (int j = 0; j < 8; ++j) {
        int c = tid + j * 256;
        int r = c >> 4, q = c & 15;   // 16 chunks per 128-elem row
        async_load_b128(qbase + c * 16,
                        Q + (rowbase + qt * 128 + r) * HDIM + col0 + q * 8);
    }

    v8f o[8];
#pragma unroll
    for (int j = 0; j < 8; ++j) o[j] = v8f_zero();
    float m_i[8], l_i[8];
#pragma unroll
    for (int r = 0; r < 8; ++r) { m_i[r] = NEG_INF; l_i[r] = 0.f; }

    for (int kt = 0; kt <= qt; ++kt) {
        __syncthreads();   // everyone done with previous Ks/Vt
        // K tile (t, d): async
#pragma unroll
        for (int j = 0; j < 8; ++j) {
            int c = tid + j * 256;
            int r = c >> 4, q = c & 15;
            async_load_b128(kbase + c * 16,
                            Kmat + (rowbase + kt * 128 + r) * HDIM + col0 + q * 8);
        }
        // V tile transposed -> (d, t): element-wise
        for (int i = tid; i < 16384; i += 256) {
            int t = i >> 7, d = i & 127;
            Vt[d * 128 + t] = V[(rowbase + kt * 128 + t) * HDIM + col0 + d];
        }
        wait_async_0();    // Q (first iter) + K present in this wave's view
        __syncthreads();   // + everyone's Vt stores and asyncs

        // scores: S = Q * K^T  (16 rows x 128 cols per wave)
        v8f s[8];
#pragma unroll
        for (int j = 0; j < 8; ++j) s[j] = v8f_zero();
#pragma unroll
        for (int dc = 0; dc < 4; ++dc) {
            const int d0 = dc * 32;
            FragB16 qa;
            {
                int mrow = wid * 16 + l16;
#pragma unroll
                for (int v = 0; v < 8; ++v) {
                    int kb = d0 + hl * 16 + 2 * v;
                    qa.u[v] = *(const unsigned int*)&Qs[mrow * 128 + kb];
                }
            }
#pragma unroll
            for (int j = 0; j < 8; ++j) {
                FragB16 kb_;
                int t = j * 16 + l16;
#pragma unroll
                for (int v = 0; v < 8; ++v) {
                    int kb = d0 + hl * 16 + 2 * v;
                    kb_.u[v] = *(const unsigned int*)&Ks[t * 128 + kb];
                }
                s[j] = __builtin_amdgcn_wmma_f32_16x16x32_bf16(
                    false, qa.v, false, kb_.v, (short)0, s[j], false, false);
            }
        }

        // scale + causal mask (only the diagonal tile needs masking)
        const bool diag = (kt == qt);
#pragma unroll
        for (int j = 0; j < 8; ++j)
#pragma unroll
            for (int r = 0; r < 8; ++r) {
                float sv = s[j][r] * scale;
                if (diag) {
                    int qrow = qt * 128 + wid * 16 + r + 8 * hl;
                    int kcol = kt * 128 + j * 16 + l16;
                    if (kcol > qrow) sv = NEG_INF;
                }
                s[j][r] = sv;
            }

        // online softmax, per fragment-row r (row = 16*wid + r + 8*hl)
#pragma unroll
        for (int r = 0; r < 8; ++r) {
            float mx = s[0][r];
#pragma unroll
            for (int j = 1; j < 8; ++j) mx = fmaxf(mx, s[j][r]);
            mx = fmaxf(mx, __shfl_xor(mx, 1));
            mx = fmaxf(mx, __shfl_xor(mx, 2));
            mx = fmaxf(mx, __shfl_xor(mx, 4));
            mx = fmaxf(mx, __shfl_xor(mx, 8));
            float mn = fmaxf(m_i[r], mx);
            float alpha = __expf(m_i[r] - mn);
            float rs = 0.f;
#pragma unroll
            for (int j = 0; j < 8; ++j) {
                float p = __expf(s[j][r] - mn);
                s[j][r] = p;
                rs += p;
            }
            rs += __shfl_xor(rs, 1);
            rs += __shfl_xor(rs, 2);
            rs += __shfl_xor(rs, 4);
            rs += __shfl_xor(rs, 8);
            l_i[r] = l_i[r] * alpha + rs;
            m_i[r] = mn;
#pragma unroll
            for (int jd = 0; jd < 8; ++jd) o[jd][r] *= alpha;
        }

        // P -> per-wave LDS slab (C-layout -> memory), then re-gather as A frags
#pragma unroll
        for (int j = 0; j < 8; ++j)
#pragma unroll
            for (int r = 0; r < 8; ++r)
                Ps[wid * 2048 + (r + 8 * hl) * 128 + j * 16 + l16] = (bf16_t)s[j][r];
        // per-wave region; LDS ops are in-order within a wave -> no barrier

        // O += P * V
#pragma unroll
        for (int tc = 0; tc < 4; ++tc) {
            const int t0 = tc * 32;
            FragB16 pa;
#pragma unroll
            for (int v = 0; v < 8; ++v) {
                int kb = t0 + hl * 16 + 2 * v;
                pa.u[v] = *(const unsigned int*)&Ps[wid * 2048 + l16 * 128 + kb];
            }
#pragma unroll
            for (int jd = 0; jd < 8; ++jd) {
                FragB16 vb;
                int d = jd * 16 + l16;
#pragma unroll
                for (int v = 0; v < 8; ++v) {
                    int kb = t0 + hl * 16 + 2 * v;
                    vb.u[v] = *(const unsigned int*)&Vt[d * 128 + kb];
                }
                o[jd] = __builtin_amdgcn_wmma_f32_16x16x32_bf16(
                    false, pa.v, false, vb.v, (short)0, o[jd], false, false);
            }
        }
    }

    // normalize and write (b, s, h*HD + d) as bf16 for the Wo GEMM
#pragma unroll
    for (int jd = 0; jd < 8; ++jd)
#pragma unroll
        for (int r = 0; r < 8; ++r) {
            float val = o[jd][r] / l_i[r];
            size_t grow = rowbase + qt * 128 + wid * 16 + r + 8 * hl;
            int gcol = col0 + jd * 16 + l16;
            O[grow * HDIM + gcol] = (bf16_t)val;
        }
}

// ---------------------------------------------------------------------------
// SwiGLU: h[r,c] = silu(gv[r,c]) * gv[r, I + c]  -> bf16
// ---------------------------------------------------------------------------
__global__ void silu_mul_kernel(const float* __restrict__ gv,
                                bf16_t* __restrict__ h) {
    long i = (long)blockIdx.x * blockDim.x + threadIdx.x;
    const long n = (long)ROWS * IDIM;
    long stride = (long)gridDim.x * blockDim.x;
    for (; i < n; i += stride) {
        long r = i / IDIM, c = i - r * IDIM;
        float g = gv[r * (2L * IDIM) + c];
        float val = gv[r * (2L * IDIM) + IDIM + c];
        float sig = 1.f / (1.f + __expf(-g));
        h[i] = (bf16_t)(g * sig * val);
    }
}

// ---------------------------------------------------------------------------
// Launch
// ---------------------------------------------------------------------------
extern "C" void kernel_launch(void* const* d_in, const int* in_sizes, int n_in,
                              void* d_out, int out_size, void* d_ws, size_t ws_size,
                              hipStream_t stream) {
    const float* x  = (const float*)d_in[0];
    const float* g1 = (const float*)d_in[1];
    const float* g2 = (const float*)d_in[2];
    const float* g3 = (const float*)d_in[3];
    const float* g4 = (const float*)d_in[4];
    const float* Wq = (const float*)d_in[5];
    const float* Wk = (const float*)d_in[6];
    const float* Wv = (const float*)d_in[7];
    const float* Wo = (const float*)d_in[8];
    const float* W1 = (const float*)d_in[9];
    const float* W2 = (const float*)d_in[10];

    char* ws = (char*)d_ws;
    size_t off = 0;
    auto alloc = [&](size_t bytes) -> char* {
        char* p = ws + off;
        off += (bytes + 255) & ~(size_t)255;
        return p;
    };

    bf16_t* Wq_b = (bf16_t*)alloc((size_t)HDIM * HDIM * 2);
    bf16_t* Wk_b = (bf16_t*)alloc((size_t)HDIM * HDIM * 2);
    bf16_t* Wv_b = (bf16_t*)alloc((size_t)HDIM * HDIM * 2);
    bf16_t* Wo_b = (bf16_t*)alloc((size_t)HDIM * HDIM * 2);
    bf16_t* W1_b = (bf16_t*)alloc((size_t)2 * IDIM * HDIM * 2);
    bf16_t* W2_b = (bf16_t*)alloc((size_t)HDIM * IDIM * 2);
    bf16_t* xn_b   = (bf16_t*)alloc((size_t)ROWS * HDIM * 2);
    bf16_t* q_b    = (bf16_t*)alloc((size_t)ROWS * HDIM * 2);
    bf16_t* k_b    = (bf16_t*)alloc((size_t)ROWS * HDIM * 2);
    bf16_t* v_b    = (bf16_t*)alloc((size_t)ROWS * HDIM * 2);
    bf16_t* attn_b = (bf16_t*)alloc((size_t)ROWS * HDIM * 2);
    float*  attnproj = (float*)alloc((size_t)ROWS * HDIM * 4);
    float*  x2       = (float*)alloc((size_t)ROWS * HDIM * 4);
    bf16_t* xn2_b  = (bf16_t*)alloc((size_t)ROWS * HDIM * 2);
    float*  gv       = (float*)alloc((size_t)ROWS * 2 * IDIM * 4);
    bf16_t* h_b    = (bf16_t*)alloc((size_t)ROWS * IDIM * 2);
    float*  mlp      = (float*)alloc((size_t)ROWS * HDIM * 4);

    // 1. weights -> bf16
    f32_to_bf16_kernel<<<2048, 256, 0, stream>>>(Wq, Wq_b, (long)HDIM * HDIM);
    f32_to_bf16_kernel<<<2048, 256, 0, stream>>>(Wk, Wk_b, (long)HDIM * HDIM);
    f32_to_bf16_kernel<<<2048, 256, 0, stream>>>(Wv, Wv_b, (long)HDIM * HDIM);
    f32_to_bf16_kernel<<<2048, 256, 0, stream>>>(Wo, Wo_b, (long)HDIM * HDIM);
    f32_to_bf16_kernel<<<4096, 256, 0, stream>>>(W1, W1_b, (long)2 * IDIM * HDIM);
    f32_to_bf16_kernel<<<4096, 256, 0, stream>>>(W2, W2_b, (long)HDIM * IDIM);

    // 2. xn = rmsnorm(x, g1) -> bf16
    rmsnorm_chain_kernel<<<ROWS, 256, 0, stream>>>(x, nullptr, g1, nullptr,
                                                   nullptr, xn_b);

    // 3. Q/K/V projections
    dim3 g_proj(HDIM / 128, ROWS / 128);
    gemm_bf16_kernel<true><<<g_proj, 256, 0, stream>>>(xn_b, Wq_b, q_b, ROWS, HDIM, HDIM);
    gemm_bf16_kernel<true><<<g_proj, 256, 0, stream>>>(xn_b, Wk_b, k_b, ROWS, HDIM, HDIM);
    gemm_bf16_kernel<true><<<g_proj, 256, 0, stream>>>(xn_b, Wv_b, v_b, ROWS, HDIM, HDIM);

    // 4. causal flash attention
    dim3 g_att(SEQ / 128, NHEADS, BATCH);
    attn_kernel<<<g_att, 256, 131072, stream>>>(q_b, k_b, v_b, attn_b);

    // 5. output projection
    gemm_bf16_kernel<false><<<g_proj, 256, 0, stream>>>(attn_b, Wo_b, attnproj,
                                                        ROWS, HDIM, HDIM);

    // 6. x2 = rmsnorm(x + attnproj, g2) [fp32]; xn2 = rmsnorm(x2, g3) [bf16]
    rmsnorm_chain_kernel<<<ROWS, 256, 0, stream>>>(x, attnproj, g2, g3, x2, xn2_b);

    // 7. MLP up (gate|val)
    dim3 g_m1((2 * IDIM) / 128, ROWS / 128);
    gemm_bf16_kernel<false><<<g_m1, 256, 0, stream>>>(xn2_b, W1_b, gv,
                                                      ROWS, 2 * IDIM, HDIM);

    // 8. SwiGLU
    silu_mul_kernel<<<4096, 256, 0, stream>>>(gv, h_b);

    // 9. MLP down
    dim3 g_m2(HDIM / 128, ROWS / 128);
    gemm_bf16_kernel<false><<<g_m2, 256, 0, stream>>>(h_b, W2_b, mlp,
                                                      ROWS, HDIM, IDIM);

    // 10. out = rmsnorm(x2 + mlp, g4) -> fp32
    rmsnorm_chain_kernel<<<ROWS, 256, 0, stream>>>(x2, mlp, g4, nullptr,
                                                   (float*)d_out, nullptr);
}
